// SparseCoder_74028056313934
// MI455X (gfx1250) — matrix operand order repaired and checked by reading, not compiled, not run on hardware
//
#include <hip/hip_runtime.h>

// ---------- types ----------
typedef __attribute__((ext_vector_type(16))) __bf16 v16bf;
typedef __attribute__((ext_vector_type(8)))  __bf16 v8bf;
typedef __attribute__((ext_vector_type(8)))  float  v8f;

union AFrag { v16bf v; v8bf h[2]; };

// ---------- problem constants ----------
#define BATCH 4096
#define DIM   1024
#define LAT   32768
#define TOPK  32
#define MROWS 32        // rows per workgroup (each wave computes both 16-row M-tiles)
#define CHUNK 256       // columns per chunk (8 N-waves x 2 subtiles x 16)
#define LPAD  1032      // padded LDS row length (bf16), keeps 16B align, kills bank conflicts

// ---------- init workspace accumulators ----------
__global__ void init_ws_k(float* colsum, float* scal) {
    int i = blockIdx.x * 256 + threadIdx.x;
    if (i < DIM) colsum[i] = 0.f;
    if (i < 2)   scal[i] = 0.f;
}

// ---------- convert x - b_dec -> bf16 ----------
__global__ void conv_x_k(const float* __restrict__ x, const float* __restrict__ b_dec,
                         __bf16* __restrict__ Xb, int n) {
    int i = blockIdx.x * 256 + threadIdx.x;
    if (i < n) Xb[i] = (__bf16)(x[i] - b_dec[i & (DIM - 1)]);
}

// ---------- convert W_enc -> bf16 ----------
__global__ void conv_w_k(const float* __restrict__ W, __bf16* __restrict__ Wb, int n) {
    int i = blockIdx.x * 256 + threadIdx.x;
    if (i < n) Wb[i] = (__bf16)W[i];
}

// ---------- per-column sums and global sum of squares of x ----------
__global__ void colstats_k(const float* __restrict__ x, float* __restrict__ colsum,
                           float* __restrict__ scal) {
    __shared__ float red[256];
    int tid = threadIdx.x;
    int r0 = blockIdx.x * 128;
    float cs0 = 0.f, cs1 = 0.f, cs2 = 0.f, cs3 = 0.f, ss = 0.f;
    for (int r = r0; r < r0 + 128; ++r) {
        const float* xr = x + (size_t)r * DIM;
        float v0 = xr[tid];       float v1 = xr[tid + 256];
        float v2 = xr[tid + 512]; float v3 = xr[tid + 768];
        cs0 += v0; cs1 += v1; cs2 += v2; cs3 += v3;
        ss += v0 * v0 + v1 * v1 + v2 * v2 + v3 * v3;
    }
    atomicAdd(&colsum[tid],       cs0);
    atomicAdd(&colsum[tid + 256], cs1);
    atomicAdd(&colsum[tid + 512], cs2);
    atomicAdd(&colsum[tid + 768], cs3);
    red[tid] = ss; __syncthreads();
    for (int off = 128; off > 0; off >>= 1) {
        if (tid < off) red[tid] += red[tid + off];
        __syncthreads();
    }
    if (tid == 0) atomicAdd(&scal[0], red[0]);
}

// ---------- fused encoder GEMM (bf16 WMMA, 2Mx2N per wave) + exact running top-32 ----------
__launch_bounds__(256)
__global__ void enc_topk_k(const __bf16* __restrict__ Xb, const __bf16* __restrict__ Wb,
                           const float* __restrict__ b_enc,
                           float* __restrict__ out_acts, float* __restrict__ out_idxf,
                           int* __restrict__ out_idxi) {
    __shared__ __bf16 Ash[MROWS * LPAD];              // 66,048 B
    __shared__ float  cval[MROWS][CHUNK];             // 32,768 B
    __shared__ int    cidx[MROWS][CHUNK];             // 32,768 B
    __shared__ float  topv[MROWS][TOPK];              //  4,096 B
    __shared__ int    topi[MROWS][TOPK];              //  4,096 B
    __shared__ __align__(16) float thr[MROWS];
    __shared__ int    ccnt[MROWS];
    __shared__ int    minpos[MROWS];

    const int tid  = threadIdx.x;
    const int wave = tid >> 5;
    const int lane = tid & 31;
    const int row0 = blockIdx.x * MROWS;

    // Stage A (32 rows x 1024 bf16) into LDS with 16B vector stores.
    for (int rep = 0; rep < 16; ++rep) {
        int linear = tid + rep * 256;                 // 0..4095 vectors of 8 bf16
        int r  = linear >> 7;                         // 128 vectors per row
        int kc = (linear & 127) * 8;
        *(v8bf*)&Ash[r * LPAD + kc] = *(const v8bf*)&Xb[(size_t)(row0 + r) * DIM + kc];
    }
    if (tid < MROWS) { thr[tid] = 0.f; ccnt[tid] = 0; minpos[tid] = 0; }
    { int r = tid >> 3, j0 = (tid & 7) * 4;           // 32 rows x 32 entries, 4 per thread
      #pragma unroll
      for (int q = 0; q < 4; ++q) { topv[r][j0 + q] = 0.f; topi[r][j0 + q] = j0 + q; }
    }
    __syncthreads();

    const int nt    = wave;               // N group 0..7 (cols nt*32..+31)
    const int khalf = lane >> 4;          // K-half select per WMMA layout
    const int lcol  = lane & 15;
    const __bf16* arow0 = &Ash[lcol * LPAD];          // M-tile 0 (rows 0..15)
    const __bf16* arow1 = &Ash[(16 + lcol) * LPAD];   // M-tile 1 (rows 16..31)
    const int rb0 = khalf * 8;            // C rows held by this lane, M-tile 0
    const int rb1 = 16 + khalf * 8;       // C rows held by this lane, M-tile 1

    for (int chunk = 0; chunk < LAT / CHUNK; ++chunk) {
        const int gcol0 = chunk * CHUNK + nt * 32 + lcol;
        const int gcol1 = gcol0 + 16;
        const __bf16* bbase0 = Wb + (size_t)gcol0 * DIM;
        const __bf16* bbase1 = Wb + (size_t)gcol1 * DIM;
        const float be0 = b_enc[gcol0];
        const float be1 = b_enc[gcol1];

        // Snapshot thresholds for this lane's 16 rows (contiguous float4 LDS loads).
        float4 t0 = *(const float4*)&thr[rb0];
        float4 t1 = *(const float4*)&thr[rb0 + 4];
        float4 t2 = *(const float4*)&thr[rb1];
        float4 t3 = *(const float4*)&thr[rb1 + 4];
        float tr0[8] = {t0.x, t0.y, t0.z, t0.w, t1.x, t1.y, t1.z, t1.w};
        float tr1[8] = {t2.x, t2.y, t2.z, t2.w, t3.x, t3.y, t3.z, t3.w};

        v8f c00 = {0.f, 0.f, 0.f, 0.f, 0.f, 0.f, 0.f, 0.f};
        v8f c01 = c00, c10 = c00, c11 = c00;
        #pragma unroll 4
        for (int k0 = 0; k0 < DIM; k0 += 32) {
            AFrag a0, a1;
            // A 16x32 bf16, contiguous k order chosen here (B matches, not the ISA table): lanes 0-15 -> K 0..15, lanes 16-31 -> K 16..31
            a0.h[0] = *(const v8bf*)(arow0 + k0 + khalf * 16);
            a0.h[1] = *(const v8bf*)(arow0 + k0 + 8 + khalf * 16);
            a1.h[0] = *(const v8bf*)(arow1 + k0 + khalf * 16);
            a1.h[1] = *(const v8bf*)(arow1 + k0 + 8 + khalf * 16);
            // B 32x16 bf16, same contiguous k order as A (not the ISA table): lane holds its column; lanes 0-15 K 0..15, lanes 16-31 K 16..31
            v16bf b0 = *(const v16bf*)(bbase0 + k0 + khalf * 16);
            v16bf b1 = *(const v16bf*)(bbase1 + k0 + khalf * 16);
            c00 = __builtin_amdgcn_wmma_f32_16x16x32_bf16(false, a0.v, false, b0,
                                                          (short)0, c00, false, false);
            c10 = __builtin_amdgcn_wmma_f32_16x16x32_bf16(false, a1.v, false, b0,
                                                          (short)0, c10, false, false);
            c01 = __builtin_amdgcn_wmma_f32_16x16x32_bf16(false, a0.v, false, b1,
                                                          (short)0, c01, false, false);
            c11 = __builtin_amdgcn_wmma_f32_16x16x32_bf16(false, a1.v, false, b1,
                                                          (short)0, c11, false, false);
        }

        // ReLU + bias; survivors above the running 32nd-max join the per-row candidate list.
        #define PUSH(VAL, RLOC, THR, GCOL)                                     \
            { float _v = (VAL);                                                \
              if (_v > 0.f && _v > (THR)) {                                    \
                  int _p = atomicAdd(&ccnt[(RLOC)], 1);                        \
                  cval[(RLOC)][_p] = _v;                                       \
                  cidx[(RLOC)][_p] = (GCOL);                                   \
              } }
        #pragma unroll
        for (int i = 0; i < 8; ++i) {
            PUSH(c00[i] + be0, rb0 + i, tr0[i], gcol0);
            PUSH(c01[i] + be1, rb0 + i, tr0[i], gcol1);
            PUSH(c10[i] + be0, rb1 + i, tr1[i], gcol0);
            PUSH(c11[i] + be1, rb1 + i, tr1[i], gcol1);
        }
        #undef PUSH
        __syncthreads();

        // Drain: one lane per row maintains exact top-32 + threshold.
        if (tid < MROWS) {
            int r = tid;
            int cnt = ccnt[r];
            float mn = thr[r]; int mp = minpos[r];
            for (int p = 0; p < cnt; ++p) {
                float v = cval[r][p];
                if (v > mn) {
                    topv[r][mp] = v; topi[r][mp] = cidx[r][p];
                    mn = topv[r][0]; mp = 0;
                    #pragma unroll
                    for (int j = 1; j < TOPK; ++j) {
                        float tv = topv[r][j];
                        if (tv < mn) { mn = tv; mp = j; }
                    }
                }
            }
            thr[r] = mn; minpos[r] = mp; ccnt[r] = 0;
        }
        __syncthreads();
    }

    // Sort descending (selection sort) and emit.
    if (tid < MROWS) {
        int r = tid;
        for (int a = 0; a < TOPK - 1; ++a) {
            int best = a; float bv = topv[r][a];
            for (int b = a + 1; b < TOPK; ++b)
                if (topv[r][b] > bv) { bv = topv[r][b]; best = b; }
            float tv = topv[r][a]; topv[r][a] = topv[r][best]; topv[r][best] = tv;
            int   ti = topi[r][a]; topi[r][a] = topi[r][best]; topi[r][best] = ti;
        }
        size_t grow = (size_t)(row0 + r);
        for (int j = 0; j < TOPK; ++j) {
            out_acts[grow * TOPK + j] = topv[r][j];
            out_idxf[grow * TOPK + j] = (float)topi[r][j];
            out_idxi[grow * TOPK + j] = topi[r][j];
        }
    }
}

// ---------- sparse decode + error accumulation ----------
__global__ void decode_k(const float* __restrict__ x, const float* __restrict__ W_dec,
                         const float* __restrict__ b_dec, const float* __restrict__ acts,
                         const int* __restrict__ idx, float* __restrict__ out,
                         float* __restrict__ err_accum) {
    __shared__ float sa[TOPK];
    __shared__ int   si[TOPK];
    __shared__ float red[256];
    const int row = blockIdx.x, tid = threadIdx.x;
    if (tid < TOPK) { sa[tid] = acts[(size_t)row * TOPK + tid]; si[tid] = idx[(size_t)row * TOPK + tid]; }
    __syncthreads();
    const int d0 = tid * 4;
    float4 acc = *(const float4*)&b_dec[d0];
    #pragma unroll 8
    for (int j = 0; j < TOPK; ++j) {
        float a = sa[j];
        float4 w = *(const float4*)&W_dec[(size_t)si[j] * DIM + d0];
        acc.x += a * w.x; acc.y += a * w.y; acc.z += a * w.z; acc.w += a * w.w;
    }
    *(float4*)&out[(size_t)row * DIM + d0] = acc;
    float4 xv = *(const float4*)&x[(size_t)row * DIM + d0];
    float ex = xv.x - acc.x, ey = xv.y - acc.y, ez = xv.z - acc.z, ew = xv.w - acc.w;
    red[tid] = ex * ex + ey * ey + ez * ez + ew * ew;
    __syncthreads();
    for (int off = 128; off > 0; off >>= 1) {
        if (tid < off) red[tid] += red[tid + off];
        __syncthreads();
    }
    if (tid == 0) atomicAdd(err_accum, red[0]);
}

// ---------- final FVU ----------
__global__ void fvu_k(const float* __restrict__ colsum, const float* __restrict__ scal,
                      float* __restrict__ out_fvu) {
    __shared__ float red[256];
    int tid = threadIdx.x;
    float p = 0.f;
    for (int d = tid; d < DIM; d += 256) { float m = colsum[d]; p += m * m; }
    red[tid] = p; __syncthreads();
    for (int off = 128; off > 0; off >>= 1) {
        if (tid < off) red[tid] += red[tid + off];
        __syncthreads();
    }
    if (tid == 0) {
        // sum((x - mean)^2) = sum(x^2) - sum_d colsum_d^2 / B
        float tv = scal[0] - red[0] / (float)BATCH;
        tv = fmaxf(tv, 1e-8f);
        out_fvu[0] = scal[1] / tv;
    }
}

// ---------- launch ----------
extern "C" void kernel_launch(void* const* d_in, const int* in_sizes, int n_in,
                              void* d_out, int out_size, void* d_ws, size_t ws_size,
                              hipStream_t stream) {
    (void)in_sizes; (void)n_in; (void)out_size; (void)ws_size;
    const float* x     = (const float*)d_in[0];
    const float* W_enc = (const float*)d_in[1];
    const float* b_enc = (const float*)d_in[2];
    const float* W_dec = (const float*)d_in[3];
    const float* b_dec = (const float*)d_in[4];
    // d_in[5] = k (known 32 at compile time)

    float* out      = (float*)d_out;
    float* out_acts = out + (size_t)BATCH * DIM;                 // 4,194,304
    float* out_idxf = out_acts + (size_t)BATCH * TOPK;           // +131,072
    float* out_fvu  = out_idxf + (size_t)BATCH * TOPK;           // +131,072 -> scalar

    char* ws = (char*)d_ws;
    __bf16* Xb     = (__bf16*)(ws);                               // 8 MB
    __bf16* Wb     = (__bf16*)(ws + (size_t)8 * 1024 * 1024);     // 64 MB
    float*  colsum = (float*)(ws + (size_t)75497472);             // 4 KB
    float*  scal   = (float*)(ws + (size_t)75501568);             // [0]=sum(x^2), [1]=sum(e^2)
    int*    idxi   = (int*)(ws + (size_t)75501824);               // 512 KB int top-indices

    init_ws_k<<<4, 256, 0, stream>>>(colsum, scal);
    conv_x_k<<<(BATCH * DIM) / 256, 256, 0, stream>>>(x, b_dec, Xb, BATCH * DIM);
    conv_w_k<<<(LAT * DIM) / 256, 256, 0, stream>>>(W_enc, Wb, LAT * DIM);
    colstats_k<<<BATCH / 128, 256, 0, stream>>>(x, colsum, scal);
    enc_topk_k<<<BATCH / MROWS, 256, 0, stream>>>(Xb, Wb, b_enc, out_acts, out_idxf, idxi);
    decode_k<<<BATCH, 256, 0, stream>>>(x, W_dec, b_dec, out_acts, idxi, out, scal + 1);
    fvu_k<<<1, 256, 0, stream>>>(colsum, scal, out_fvu);
}
